// KANGaussianPolicy_29274497089813
// MI455X (gfx1250) — matrix operand order repaired and checked by reading, not compiled, not run on hardware
//
#include <hip/hip_runtime.h>
#include <hip/hip_bf16.h>
#include <math.h>

// ---------------- types ----------------
typedef __attribute__((ext_vector_type(16))) __bf16 bf16x16;
typedef __attribute__((ext_vector_type(8)))  float  f32x8;

union FragAB { bf16x16 bf; uint4 q[2]; };

// ---------------- constants ----------------
#define B_TOTAL 32768
#define S_IN    256
#define H_MID   512
#define A_OUT   64
#define NB      8

#define K1      2304              // 256 silu + 256*8 spline
#define AROW1   2312              // padded bf16 row stride (1156 dwords, %64==4)
#define NT1     32                // 512/16
#define KT1     72                // 2304/32
#define ABYTES1 (32 * AROW1 * 2)  // 147968
#define BSLAB1  (NT1 * 512 * 2)   // 32768 B per kt
#define SMEM1   (ABYTES1 + 2 * BSLAB1)   // 213504

#define K2      4608              // 512 silu + 512*8 spline
#define AROW2   4616              // padded (2308 dwords, %64==4)
#define NT2     8                 // 128/16
#define KT2     144               // 4608/32
#define ABYTES2 (32 * AROW2 * 2)  // 295424
#define BSLAB2  (NT2 * 512 * 2)   // 8192 B per kt
#define SMEM2   (ABYTES2 + 2 * BSLAB2)   // 311808

// ---------------- helpers ----------------
__device__ __forceinline__ unsigned short f2bf(float f) {
  unsigned u = __builtin_bit_cast(unsigned, f);
  u += 0x7FFFu + ((u >> 16) & 1u);   // round-to-nearest-even
  return (unsigned short)(u >> 16);
}

// Cox-de Boor cubic B-spline bases on uniform knots t_j = -2.2 + 0.4*j (12 knots)
__device__ __forceinline__ void bspline8(float x, float* out) {
  float b[11];
#pragma unroll
  for (int j = 0; j < 11; ++j) {
    float t0 = -2.2f + 0.4f * j;
    b[j] = (x >= t0 && x < t0 + 0.4f) ? 1.0f : 0.0f;
  }
#pragma unroll
  for (int p = 1; p <= 3; ++p) {
    float inv = 1.0f / (0.4f * (float)p);
#pragma unroll
    for (int j = 0; j < 11 - p; ++j) {
      float tj   = -2.2f + 0.4f * j;
      float tjp1 = tj + 0.4f * (float)(p + 1);
      b[j] = (x - tj) * inv * b[j] + (tjp1 - x) * inv * b[j + 1];
    }
  }
#pragma unroll
  for (int j = 0; j < 8; ++j) out[j] = b[j];
}

__device__ __forceinline__ float silu_f(float x) {
  return x / (1.0f + __expf(-x));
}

// CDNA5 async global->LDS copy (16B per lane), tracked by ASYNCcnt
__device__ __forceinline__ void async_copy16(unsigned lds_off, const void* gaddr) {
  unsigned long long a = (unsigned long long)gaddr;
  asm volatile("global_load_async_to_lds_b128 %0, %1, off"
               :: "v"(lds_off), "v"(a) : "memory");
}
__device__ __forceinline__ void wait_async0() {
  asm volatile("s_wait_asynccnt 0x0" ::: "memory");
}

// Cooperative slab memcpy: nbytes (multiple of 4096) from g to LDS, 256 threads
template <int NBYTES>
__device__ __forceinline__ void fill_slab(unsigned lds_base, const void* g, int tid) {
#pragma unroll
  for (int i = 0; i < NBYTES / 4096; ++i) {
    int off = (i * 256 + tid) * 16;
    async_copy16(lds_base + off, (const char*)g + off);
  }
}

// ---------------- weight repack: f32 -> bf16, WMMA-fragment-tiled ----------------
// Tile (kt, nt) is 32x16 (KxN), 1024 B. Within a tile: chunk c (0/1), lane (0..31),
// jj (0..7): bf16 at [c*256 + lane*8 + jj] = W[k][n],
//   n = nt*16 + lane%16,  k = kt*32 + c*16 + (lane>>4)*8 + jj.
// A lane's fragment = 16B at lane*16 plus 16B at 512 + lane*16 (conflict-friendly).
__global__ void repack_w1_kernel(const float* __restrict__ base_w,
                                 const float* __restrict__ spline_w,
                                 unsigned short* __restrict__ dst) {
  int e = blockIdx.x * 256 + threadIdx.x;
  if (e >= K1 * H_MID) return;
  int jj   = e & 7;
  int l    = (e >> 3) & 31;
  int c    = (e >> 8) & 1;
  int tile = e >> 9;
  int nt   = tile & (NT1 - 1);
  int kt   = tile / NT1;
  int n    = nt * 16 + (l & 15);
  int k    = kt * 32 + c * 16 + (l >> 4) * 8 + jj;
  float v;
  if (k < S_IN) v = base_w[n * S_IN + k];
  else          v = spline_w[n * (S_IN * NB) + (k - S_IN)]; // [o][i][g] contiguous
  dst[e] = f2bf(v);
}

__global__ void repack_w2_kernel(const float* __restrict__ mean_bw,
                                 const float* __restrict__ mean_sw,
                                 const float* __restrict__ lstd_bw,
                                 const float* __restrict__ lstd_sw,
                                 unsigned short* __restrict__ dst) {
  int e = blockIdx.x * 256 + threadIdx.x;
  if (e >= K2 * 2 * A_OUT) return;
  int jj   = e & 7;
  int l    = (e >> 3) & 31;
  int c    = (e >> 8) & 1;
  int tile = e >> 9;
  int nt   = tile & (NT2 - 1);
  int kt   = tile / NT2;
  int n    = nt * 16 + (l & 15);            // 0..127 (mean | lstd)
  int k    = kt * 32 + c * 16 + (l >> 4) * 8 + jj;
  const float* bw = (n < A_OUT) ? mean_bw : lstd_bw;
  const float* sw = (n < A_OUT) ? mean_sw : lstd_sw;
  int nn = n & (A_OUT - 1);
  float v;
  if (k < H_MID) v = bw[nn * H_MID + k];
  else           v = sw[nn * (H_MID * NB) + (k - H_MID)];
  dst[e] = f2bf(v);
}

// ---------------- layer 1: KAN feature layer + LayerNorm + ReLU ----------------
__global__ __launch_bounds__(256)
void kan_feat_kernel(const float* __restrict__ state,
                     const unsigned short* __restrict__ W1,
                     const float* __restrict__ gamma,
                     const float* __restrict__ beta,
                     float* __restrict__ xmid) {
  extern __shared__ __align__(16) char smem[];
  unsigned short* as = (unsigned short*)smem;
  const int tid  = threadIdx.x;
  const int lane = tid & 31;
  const int wave = tid >> 5;
  const int row0 = blockIdx.x * 32;
  const unsigned bbase = (unsigned)(unsigned long long)(smem + ABYTES1);

  // Kick off async staging of the first weight slab while we compute activations
  fill_slab<BSLAB1>(bbase, W1, tid);

  // Phase 1: build activation block [32 rows][2304] bf16 in LDS
#pragma unroll 2
  for (int r = 0; r < 32; ++r) {
    float x  = state[(size_t)(row0 + r) * S_IN + tid];
    as[r * AROW1 + tid] = f2bf(silu_f(x));
    float b[8]; bspline8(x, b);
    unsigned* bp = (unsigned*)(as + r * AROW1 + S_IN + tid * NB);
#pragma unroll
    for (int g = 0; g < 4; ++g)
      bp[g] = (unsigned)f2bf(b[2 * g]) | ((unsigned)f2bf(b[2 * g + 1]) << 16);
  }
  __syncthreads();

  // Phase 2: WMMA GEMM. wave -> row-group (wave&1), 8 N-tiles at ((wave>>1)*8)
  const int rg   = wave & 1;
  const int nt0  = (wave >> 1) * 8;
  const int h    = lane >> 4;
  const char* arow = (const char*)(as + (size_t)(rg * 16 + (lane & 15)) * AROW1);

  f32x8 acc[8];
#pragma unroll
  for (int t = 0; t < 8; ++t)
#pragma unroll
    for (int e = 0; e < 8; ++e) acc[t][e] = 0.0f;

  for (int kt = 0; kt < KT1; ++kt) {
    const int cur = kt & 1;
    wait_async0();        // my share of slab[cur] landed
    __syncthreads();      // everyone's share landed; prev slab fully consumed
    if (kt + 1 < KT1)
      fill_slab<BSLAB1>(bbase + (cur ^ 1) * BSLAB1,
                        (const char*)W1 + (size_t)(kt + 1) * BSLAB1, tid);

    FragAB fa;
    const char* ab = arow + kt * 64 + h * 16;
    fa.q[0] = *(const uint4*)(ab);
    fa.q[1] = *(const uint4*)(ab + 32);
    const char* bslab = smem + ABYTES1 + cur * BSLAB1;
#pragma unroll
    for (int t = 0; t < 8; ++t) {
      FragAB fb;
      const char* bt = bslab + (nt0 + t) * 1024 + lane * 16;
      fb.q[0] = *(const uint4*)(bt);
      fb.q[1] = *(const uint4*)(bt + 512);
      acc[t] = __builtin_amdgcn_wmma_f32_16x16x32_bf16(
          false, fa.bf, false, fb.bf, (short)0, acc[t], false, false);
    }
  }
  __syncthreads();

  // Epilogue: stash y[32][512] (stride 516 f32) into LDS (reuse A buffer)
  float* y = (float*)smem;
#pragma unroll
  for (int t = 0; t < 8; ++t)
#pragma unroll
    for (int r = 0; r < 8; ++r) {
      int row = rg * 16 + h * 8 + r;
      int col = (nt0 + t) * 16 + (lane & 15);
      y[row * 516 + col] = acc[t][r];
    }
  __syncthreads();

  // LayerNorm reductions
  float* red = (float*)(smem + 32 * 516 * 4);  // [2][8][32] + mu[32] + rstd[32]
  {
    int rr = tid & 31, jj = tid >> 5;
    float s = 0.0f, sq = 0.0f;
    const float* yr = y + rr * 516 + jj * 64;
#pragma unroll 8
    for (int c = 0; c < 64; ++c) { float v = yr[c]; s += v; sq += v * v; }
    red[jj * 32 + rr]       = s;
    red[256 + jj * 32 + rr] = sq;
  }
  __syncthreads();
  float* mu = red + 512;
  float* rstd = red + 544;
  if (tid < 32) {
    float s = 0.0f, sq = 0.0f;
#pragma unroll
    for (int jj = 0; jj < 8; ++jj) { s += red[jj * 32 + tid]; sq += red[256 + jj * 32 + tid]; }
    float m   = s * (1.0f / 512.0f);
    float var = sq * (1.0f / 512.0f) - m * m;
    mu[tid]   = m;
    rstd[tid] = __frsqrt_rn(var + 1e-5f);
  }
  __syncthreads();

  // normalize + ReLU + coalesced store
  for (int rep = 0; rep < 64; ++rep) {
    int idx = rep * 256 + tid;
    int row = idx >> 9, col = idx & 511;
    float v = (y[row * 516 + col] - mu[row]) * rstd[row] * gamma[col] + beta[col];
    v = fmaxf(v, 0.0f);
    xmid[(size_t)(row0 + row) * H_MID + col] = v;
  }
}

// ---------------- layer 2: mean/log_std heads (32-row tile) ----------------
__global__ __launch_bounds__(256)
void kan_heads_kernel(const float* __restrict__ xmid,
                      const unsigned short* __restrict__ W2,
                      float* __restrict__ out) {
  extern __shared__ __align__(16) char smem[];
  unsigned short* as = (unsigned short*)smem;
  const int tid  = threadIdx.x;
  const int lane = tid & 31;
  const int wave = tid >> 5;
  const int row0 = blockIdx.x * 32;
  const unsigned bbase = (unsigned)(unsigned long long)(smem + ABYTES2);

  fill_slab<BSLAB2>(bbase, W2, tid);

  // Phase 1: activation block [32 rows][4608] bf16
#pragma unroll 2
  for (int rep = 0; rep < 64; ++rep) {
    int idx = rep * 256 + tid;
    int r = idx >> 9, i = idx & 511;
    float x = xmid[(size_t)(row0 + r) * H_MID + i];
    as[r * AROW2 + i] = f2bf(silu_f(x));
    float b[8]; bspline8(x, b);
    unsigned* bp = (unsigned*)(as + r * AROW2 + H_MID + i * NB);
#pragma unroll
    for (int g = 0; g < 4; ++g)
      bp[g] = (unsigned)f2bf(b[2 * g]) | ((unsigned)f2bf(b[2 * g + 1]) << 16);
  }
  __syncthreads();

  // Phase 2: wave w owns N-tile w for BOTH row groups (2 WMMA / k-step)
  const int h = lane >> 4;
  const char* arow0 = (const char*)(as + (size_t)(lane & 15) * AROW2);
  const char* arow1 = (const char*)(as + (size_t)(16 + (lane & 15)) * AROW2);
  f32x8 acc[2];
#pragma unroll
  for (int g = 0; g < 2; ++g)
#pragma unroll
    for (int e = 0; e < 8; ++e) acc[g][e] = 0.0f;

  for (int kt = 0; kt < KT2; ++kt) {
    const int cur = kt & 1;
    wait_async0();
    __syncthreads();
    if (kt + 1 < KT2)
      fill_slab<BSLAB2>(bbase + (cur ^ 1) * BSLAB2,
                        (const char*)W2 + (size_t)(kt + 1) * BSLAB2, tid);

    FragAB fb;
    const char* bt = smem + ABYTES2 + cur * BSLAB2 + wave * 1024 + lane * 16;
    fb.q[0] = *(const uint4*)(bt);
    fb.q[1] = *(const uint4*)(bt + 512);

    FragAB fa0, fa1;
    const char* a0 = arow0 + kt * 64 + h * 16;
    const char* a1 = arow1 + kt * 64 + h * 16;
    fa0.q[0] = *(const uint4*)(a0);
    fa0.q[1] = *(const uint4*)(a0 + 32);
    fa1.q[0] = *(const uint4*)(a1);
    fa1.q[1] = *(const uint4*)(a1 + 32);

    acc[0] = __builtin_amdgcn_wmma_f32_16x16x32_bf16(
        false, fa0.bf, false, fb.bf, (short)0, acc[0], false, false);
    acc[1] = __builtin_amdgcn_wmma_f32_16x16x32_bf16(
        false, fa1.bf, false, fb.bf, (short)0, acc[1], false, false);
  }

  // Epilogue: tanh(mean) | clip(log_std), straight from accumulator layout
  int colg = wave * 16 + (lane & 15);   // 0..127
#pragma unroll
  for (int g = 0; g < 2; ++g)
#pragma unroll
    for (int r = 0; r < 8; ++r) {
      int row = g * 16 + h * 8 + r;
      float v = acc[g][r];
      if (colg < A_OUT) {
        out[(size_t)(row0 + row) * A_OUT + colg] = tanhf(v);
      } else {
        v = fminf(fmaxf(v, -20.0f), 2.0f);
        out[(size_t)B_TOTAL * A_OUT + (size_t)(row0 + row) * A_OUT + (colg - A_OUT)] = v;
      }
    }
}

// ---------------- host launch ----------------
extern "C" void kernel_launch(void* const* d_in, const int* in_sizes, int n_in,
                              void* d_out, int out_size, void* d_ws, size_t ws_size,
                              hipStream_t stream) {
  (void)in_sizes; (void)n_in; (void)out_size; (void)ws_size;

  const float* state         = (const float*)d_in[0];
  const float* feat_base_w   = (const float*)d_in[1];
  const float* feat_spline_w = (const float*)d_in[2];
  const float* ln_gamma      = (const float*)d_in[3];
  const float* ln_beta       = (const float*)d_in[4];
  const float* mean_base_w   = (const float*)d_in[5];
  const float* mean_spline_w = (const float*)d_in[6];
  const float* lstd_base_w   = (const float*)d_in[7];
  const float* lstd_spline_w = (const float*)d_in[8];
  float* out = (float*)d_out;

  char* ws = (char*)d_ws;
  float*          xmid = (float*)ws;                                       // 67,108,864 B
  unsigned short* W1   = (unsigned short*)(ws + (size_t)B_TOTAL * H_MID * 4);
  unsigned short* W2   = W1 + (size_t)K1 * H_MID;

  hipFuncSetAttribute((const void*)kan_feat_kernel,
                      hipFuncAttributeMaxDynamicSharedMemorySize, SMEM1);
  hipFuncSetAttribute((const void*)kan_heads_kernel,
                      hipFuncAttributeMaxDynamicSharedMemorySize, SMEM2);

  repack_w1_kernel<<<(K1 * H_MID) / 256, 256, 0, stream>>>(
      feat_base_w, feat_spline_w, W1);
  repack_w2_kernel<<<(K2 * 2 * A_OUT) / 256, 256, 0, stream>>>(
      mean_base_w, mean_spline_w, lstd_base_w, lstd_spline_w, W2);

  kan_feat_kernel<<<B_TOTAL / 32, 256, SMEM1, stream>>>(
      state, W1, ln_gamma, ln_beta, xmid);
  kan_heads_kernel<<<B_TOTAL / 32, 256, SMEM2, stream>>>(xmid, W2, out);
}
